// CausalGCN_41721312313876
// MI455X (gfx1250) — matrix operand order repaired and checked by reading, not compiled, not run on hardware
//
#include <hip/hip_runtime.h>
#include <hip/hip_bf16.h>

// ---------------- problem constants (match reference) ----------------
#define NN 40000
#define EE 480000
#define HH 128
#define CC 10
#define GG 128

typedef __attribute__((ext_vector_type(16))) __bf16 v16bf;
typedef __attribute__((ext_vector_type(8)))  float  v8f;

union FragBf { v16bf v; uint4 q[2]; };

// ---------------------------------------------------------------------
// Column statistics -> BN fold coefficients: a = g*rstd, c = b - mean*a
// One block per column (128 blocks), biased variance (matches jnp.var).
// ---------------------------------------------------------------------
__global__ __launch_bounds__(256)
void k_colstats(const float* __restrict__ X, int rows,
                const float* __restrict__ g, const float* __restrict__ b,
                float* __restrict__ aOut, float* __restrict__ cOut) {
    __shared__ float ss[256], ss2[256];
    const int colI = blockIdx.x;
    float s = 0.f, s2 = 0.f;
    for (int r = threadIdx.x; r < rows; r += 256) {
        float v = X[r * HH + colI];
        s += v; s2 += v * v;
    }
    ss[threadIdx.x] = s; ss2[threadIdx.x] = s2;
    __syncthreads();
    for (int o = 128; o > 0; o >>= 1) {
        if (threadIdx.x < o) {
            ss[threadIdx.x]  += ss[threadIdx.x + o];
            ss2[threadIdx.x] += ss2[threadIdx.x + o];
        }
        __syncthreads();
    }
    if (threadIdx.x == 0) {
        float inv_n = 1.f / (float)rows;
        float m   = ss[0] * inv_n;
        float var = ss2[0] * inv_n - m * m;
        float rstd = rsqrtf(var + 1e-5f);
        float a = g[colI] * rstd;
        aOut[colI] = a;
        cOut[colI] = b[colI] - m * a;
    }
}

// ---------------------------------------------------------------------
// Fused BN-fold + GEMM via v_wmma_f32_16x16x32_bf16.
// Y[r][n] = act( sum_k (X[r][k]*a[k]+c[k]) * W[k][n] + bias[n] )
// Block = 256 threads = 8 waves; block tile = 128 rows x 128 cols.
// Wave w owns rows [128*bx + 16w, +16): 8 accumulators of 16x16 fp32.
// ---------------------------------------------------------------------
__global__ __launch_bounds__(256)
void k_gemm_bn_wmma(const float* __restrict__ X,
                    const float* __restrict__ W,     // [128][128] row-major (k,n)
                    const float* __restrict__ afold,
                    const float* __restrict__ cfold,
                    const float* __restrict__ bias,  // may be null
                    float* __restrict__ Y,
                    int nrows, int relu) {
    __shared__ __align__(16) __bf16 sWt[HH][HH];   // transposed: [n][k], 32 KB
    __shared__ __align__(16) __bf16 sA[HH][32];    // BN-folded A K-tile, 8 KB
    __shared__ float sAf[HH], sCf[HH];

    const int tid = threadIdx.x;
    if (tid < HH) { sAf[tid] = afold[tid]; sCf[tid] = cfold[tid]; }

    // Stage W as bf16, transposed, in LDS (once per block).
    for (int idx = tid; idx < HH * HH; idx += 256) {
        int k = idx >> 7, n = idx & 127;
        sWt[n][k] = (__bf16)W[idx];
    }
    __syncthreads();

    const int wave = tid >> 5;
    const int lane = tid & 31;
    const int half = lane >> 4;       // 0: lanes 0-15, 1: lanes 16-31
    const int l16  = lane & 15;
    const int rowBase = blockIdx.x * HH;

    v8f acc[8] = {};

    for (int ks = 0; ks < 4; ++ks) {
        // Cooperative BN-folded A tile: 128 rows x 32 K in bf16.
        for (int idx = tid; idx < HH * 32; idx += 256) {
            int r = idx >> 5, kk = idx & 31;
            int gr = rowBase + r;
            int gk = ks * 32 + kk;
            float v = 0.f;
            if (gr < nrows) v = X[gr * HH + gk] * sAf[gk] + sCf[gk];
            sA[r][kk] = (__bf16)v;
        }
        __syncthreads();

        // A fragment (16x32 bf16): lanes 0-15 hold K {0..7,16..23},
        // lanes 16-31 hold K {8..15,24..31} for row M = lane%16.
        FragBf af;
        {
            int r  = wave * 16 + l16;
            int kb = half ? 8 : 0;
            af.q[0] = *(const uint4*)&sA[r][kb];
            af.q[1] = *(const uint4*)&sA[r][kb + 16];
        }
        // 8 column tiles of B (32x16 bf16 each) from transposed-W LDS.
        for (int t = 0; t < 8; ++t) {
            FragBf bf_;
            int n  = t * 16 + l16;
            int kh = (half ? 8 : 0) + ks * 32;
            bf_.q[0] = *(const uint4*)&sWt[n][kh];
            bf_.q[1] = *(const uint4*)&sWt[n][kh + 16];
            acc[t] = __builtin_amdgcn_wmma_f32_16x16x32_bf16(
                false, af.v, false, bf_.v, (short)0, acc[t], false, false);
        }
        __syncthreads();
    }

    // Epilogue: bias + optional ReLU, fp32 store.
    for (int t = 0; t < 8; ++t) {
        int n = t * 16 + l16;
        float bv = bias ? bias[n] : 0.f;
        for (int j = 0; j < 8; ++j) {
            int m  = j + (half ? 8 : 0);
            int gr = rowBase + wave * 16 + m;
            if (gr < nrows) {
                float v = acc[t][j] + bv;
                if (relu) v = fmaxf(v, 0.f);
                Y[gr * HH + n] = v;
            }
        }
    }
}

// ---------------------------------------------------------------------
// Degree / inverse-sqrt-degree for symmetric GCN normalization.
// ---------------------------------------------------------------------
__global__ void k_deg_init(float* deg, int n) {
    int i = blockIdx.x * blockDim.x + threadIdx.x;
    if (i < n) deg[i] = 1.0f;   // self-loop weight
}
__global__ void k_deg_edges(float* deg, const int* __restrict__ row,
                            const float* __restrict__ w, int e) {
    int i = blockIdx.x * blockDim.x + threadIdx.x;
    if (i < e) atomicAdd(&deg[row[i]], w ? w[i] : 1.0f);
}
__global__ void k_dinv(const float* __restrict__ deg, float* __restrict__ dinv, int n) {
    int i = blockIdx.x * blockDim.x + threadIdx.x;
    if (i < n) dinv[i] = rsqrtf(deg[i]);
}

// ---------------------------------------------------------------------
// Propagation: out = self-loop term, then edge scatter-add.
// Wave per node/edge; lane handles a float4 feature chunk (coalesced).
// ---------------------------------------------------------------------
__global__ __launch_bounds__(256)
void k_prop_init(float* __restrict__ out, const float* __restrict__ z,
                 const float* __restrict__ dinv, int n) {
    int gid = blockIdx.x * blockDim.x + threadIdx.x;
    int i = gid >> 5;
    if (i >= n) return;
    int f = (gid & 31) * 4;
    float d = dinv[i];
    float coef = d * d;
    float4 zv = *(const float4*)(z + i * HH + f);
    float4 ov = make_float4(coef * zv.x, coef * zv.y, coef * zv.z, coef * zv.w);
    *(float4*)(out + i * HH + f) = ov;
}

__global__ __launch_bounds__(256)
void k_prop_edges(float* __restrict__ out, const float* __restrict__ z,
                  const float* __restrict__ dinv,
                  const int* __restrict__ row, const int* __restrict__ col,
                  const float* __restrict__ w, int e) {
    int gid = blockIdx.x * blockDim.x + threadIdx.x;
    int ei = gid >> 5;
    if (ei >= e) return;
    int f = (gid & 31) * 4;
    int r = row[ei], c = col[ei];
    float coef = dinv[r] * (w ? w[ei] : 1.0f) * dinv[c];
    float4 zv = *(const float4*)(z + r * HH + f);
    float* o = out + c * HH + f;
    atomicAdd(o + 0, coef * zv.x);
    atomicAdd(o + 1, coef * zv.y);
    atomicAdd(o + 2, coef * zv.z);
    atomicAdd(o + 3, coef * zv.w);
}

__global__ void k_bias_relu(float* __restrict__ dst, const float* __restrict__ src,
                            const float* __restrict__ bias, int n) {
    int gid = blockIdx.x * blockDim.x + threadIdx.x;
    if (gid < n * HH) dst[gid] = fmaxf(src[gid] + bias[gid & 127], 0.f);
}

// ---------------------------------------------------------------------
// Node attention (2-way softmax) fused with xc/xo gating. Wave per node.
// ---------------------------------------------------------------------
__global__ __launch_bounds__(256)
void k_natt_split(const float* __restrict__ h, const float* __restrict__ Wn,
                  const float* __restrict__ bn,
                  float* __restrict__ xc, float* __restrict__ xo, int n) {
    int wid  = (blockIdx.x * blockDim.x + threadIdx.x) >> 5;
    int lane = threadIdx.x & 31;
    if (wid >= n) return;                 // wave-uniform
    int f = lane * 4;
    float4 hv = *(const float4*)(h + wid * HH + f);
    float hvv[4] = { hv.x, hv.y, hv.z, hv.w };
    float s0 = 0.f, s1 = 0.f;
    for (int k = 0; k < 4; ++k) {
        s0 += hvv[k] * Wn[(f + k) * 2 + 0];
        s1 += hvv[k] * Wn[(f + k) * 2 + 1];
    }
    for (int m = 16; m > 0; m >>= 1) {
        s0 += __shfl_xor(s0, m, 32);
        s1 += __shfl_xor(s1, m, 32);
    }
    s0 += bn[0]; s1 += bn[1];
    float mx = fmaxf(s0, s1);
    float e0 = __expf(s0 - mx), e1 = __expf(s1 - mx);
    float inv = 1.f / (e0 + e1);
    float p0 = e0 * inv, p1 = e1 * inv;
    *(float4*)(xc + wid * HH + f) = make_float4(p0*hv.x, p0*hv.y, p0*hv.z, p0*hv.w);
    *(float4*)(xo + wid * HH + f) = make_float4(p1*hv.x, p1*hv.y, p1*hv.z, p1*hv.w);
}

// ---------------------------------------------------------------------
// Edge attention: softmax([h[r],h[c]] @ W_eatt + b). Wave per edge.
// ---------------------------------------------------------------------
__global__ __launch_bounds__(256)
void k_eatt(const float* __restrict__ h, const int* __restrict__ row,
            const int* __restrict__ col, const float* __restrict__ We,
            const float* __restrict__ be,
            float* __restrict__ ea0, float* __restrict__ ea1, int e) {
    int wid  = (blockIdx.x * blockDim.x + threadIdx.x) >> 5;
    int lane = threadIdx.x & 31;
    if (wid >= e) return;                 // wave-uniform
    int r = row[wid], c = col[wid];
    int f = lane * 4;
    float4 hr = *(const float4*)(h + r * HH + f);
    float4 hc = *(const float4*)(h + c * HH + f);
    float hrv[4] = { hr.x, hr.y, hr.z, hr.w };
    float hcv[4] = { hc.x, hc.y, hc.z, hc.w };
    float s0 = 0.f, s1 = 0.f;
    for (int k = 0; k < 4; ++k) {
        int fk = f + k;
        s0 += hrv[k] * We[fk * 2 + 0] + hcv[k] * We[(HH + fk) * 2 + 0];
        s1 += hrv[k] * We[fk * 2 + 1] + hcv[k] * We[(HH + fk) * 2 + 1];
    }
    for (int m = 16; m > 0; m >>= 1) {
        s0 += __shfl_xor(s0, m, 32);
        s1 += __shfl_xor(s1, m, 32);
    }
    if (lane == 0) {
        s0 += be[0]; s1 += be[1];
        float mx = fmaxf(s0, s1);
        float e0 = __expf(s0 - mx), e1 = __expf(s1 - mx);
        float inv = 1.f / (e0 + e1);
        ea0[wid] = e0 * inv;
        ea1[wid] = e1 * inv;
    }
}

// ---------------------------------------------------------------------
// Pooling / elementwise helpers.
// ---------------------------------------------------------------------
__global__ void k_zero(float* p, int n) {
    int i = blockIdx.x * blockDim.x + threadIdx.x;
    if (i < n) p[i] = 0.f;
}
__global__ __launch_bounds__(256)
void k_pool(float* __restrict__ pooled, const float* __restrict__ x,
            const int* __restrict__ batch, int n) {
    int gid = blockIdx.x * blockDim.x + threadIdx.x;
    int i = gid >> 5;
    if (i >= n) return;
    int f = (gid & 31) * 4;
    int g = batch[i];
    float4 v = *(const float4*)(x + i * HH + f);
    float* o = pooled + g * HH + f;
    atomicAdd(o + 0, v.x); atomicAdd(o + 1, v.y);
    atomicAdd(o + 2, v.z); atomicAdd(o + 3, v.w);
}
__global__ void k_add(float* __restrict__ o, const float* __restrict__ a,
                      const float* __restrict__ b, int n) {
    int i = blockIdx.x * blockDim.x + threadIdx.x;
    if (i < n) o[i] = a[i] + b[i];
}

// ---------------------------------------------------------------------
// Readout tail: BN-folded [G,128]@[128,10] + bias, then log_softmax.
// ---------------------------------------------------------------------
__global__ void k_gemm_small(const float* __restrict__ T, const float* __restrict__ W2,
                             const float* __restrict__ a2, const float* __restrict__ c2,
                             const float* __restrict__ b2, float* __restrict__ logits) {
    int gid = blockIdx.x * blockDim.x + threadIdx.x;
    if (gid >= GG * CC) return;
    int g = gid / CC, cc = gid % CC;
    float acc = b2[cc];
    for (int k = 0; k < HH; ++k)
        acc += (T[g * HH + k] * a2[k] + c2[k]) * W2[k * CC + cc];
    logits[gid] = acc;
}
__global__ void k_logsoftmax(const float* __restrict__ logits, float* __restrict__ out) {
    int g = blockIdx.x * blockDim.x + threadIdx.x;
    if (g >= GG) return;
    float v[CC];
    float mx = -1e30f;
    for (int i = 0; i < CC; ++i) { v[i] = logits[g * CC + i]; mx = fmaxf(mx, v[i]); }
    float s = 0.f;
    for (int i = 0; i < CC; ++i) s += __expf(v[i] - mx);
    float l = logf(s);
    for (int i = 0; i < CC; ++i) out[g * CC + i] = v[i] - mx - l;
}

// ---------------------------------------------------------------------
// Host orchestration.
// ---------------------------------------------------------------------
extern "C" void kernel_launch(void* const* d_in, const int* in_sizes, int n_in,
                              void* d_out, int out_size, void* d_ws, size_t ws_size,
                              hipStream_t stream) {
    const float* x        = (const float*)d_in[0];
    const int*   ei       = (const int*)  d_in[1];
    const int*   batch    = (const int*)  d_in[2];
    const float* bnf_g    = (const float*)d_in[3];
    const float* bnf_b    = (const float*)d_in[4];
    const float* W_feat   = (const float*)d_in[5];
    const float* bns_g    = (const float*)d_in[6];
    const float* bns_b    = (const float*)d_in[7];
    const float* conv_W   = (const float*)d_in[8];
    const float* conv_b   = (const float*)d_in[9];
    const float* W_eatt   = (const float*)d_in[10];
    const float* b_eatt   = (const float*)d_in[11];
    const float* W_natt   = (const float*)d_in[12];
    const float* b_natt   = (const float*)d_in[13];
    const float* bnc_g    = (const float*)d_in[14];
    const float* bnc_b    = (const float*)d_in[15];
    const float* bno_g    = (const float*)d_in[16];
    const float* bno_b    = (const float*)d_in[17];
    const float* Wc       = (const float*)d_in[18];
    const float* bc       = (const float*)d_in[19];
    const float* Wo       = (const float*)d_in[20];
    const float* bo       = (const float*)d_in[21];
    const float* ro_bn1_g = (const float*)d_in[22];
    const float* ro_bn1_b = (const float*)d_in[23];
    const float* ro_W1    = (const float*)d_in[24];
    const float* ro_b1    = (const float*)d_in[25];
    const float* ro_bn2_g = (const float*)d_in[26];
    const float* ro_bn2_b = (const float*)d_in[27];
    const float* ro_W2    = (const float*)d_in[28];
    const float* ro_b2    = (const float*)d_in[29];

    const int* row = ei;
    const int* col = ei + EE;
    float* out = (float*)d_out;

    // Workspace carve-up (floats).
    const size_t NHf = (size_t)NN * HH;
    float* base  = (float*)d_ws;
    float* hbuf  = base;
    float* zbuf  = base + NHf;
    float* pbuf  = base + 2 * NHf;
    float* deg   = base + 3 * NHf;
    float* dinvO = deg + NN;
    float* dinvA = dinvO + NN;
    float* ea0   = dinvA + NN;
    float* ea1   = ea0 + EE;
    float* xcp   = ea1 + EE;
    float* xop   = xcp + GG * HH;
    float* xsum  = xop + GG * HH;
    float* tbuf  = xsum + GG * HH;
    float* stA   = tbuf + GG * HH;
    float* stC   = stA + HH;
    float* stA2  = stC + HH;
    float* stC2  = stA2 + HH;
    float* logit = stC2 + HH;

    const int gRows = (NN + 127) / 128;            // GEMM blocks over nodes
    const int gN    = (NN + 255) / 256;
    const int gE    = (EE + 255) / 256;
    const int gNW   = (NN * 32 + 255) / 256;       // wave-per-node, lane=float4
    const int gEW   = (EE * 32 + 255) / 256;       // wave-per-edge, lane=float4
    const int gNH   = (NN * HH + 255) / 256;

    // 1) h = relu(BN(x) @ W_feat)
    k_colstats<<<HH, 256, 0, stream>>>(x, NN, bnf_g, bnf_b, stA, stC);
    k_gemm_bn_wmma<<<gRows, 256, 0, stream>>>(x, W_feat, stA, stC, nullptr, hbuf, NN, 1);

    // Shared degree normalization for the 3 unit-weight conv layers.
    k_deg_init<<<gN, 256, 0, stream>>>(deg, NN);
    k_deg_edges<<<gE, 256, 0, stream>>>(deg, row, nullptr, EE);
    k_dinv<<<gN, 256, 0, stream>>>(deg, dinvO, NN);

    // 2) three GCN conv layers
    for (int i = 0; i < 3; ++i) {
        k_colstats<<<HH, 256, 0, stream>>>(hbuf, NN, bns_g + i * HH, bns_b + i * HH, stA, stC);
        k_gemm_bn_wmma<<<gRows, 256, 0, stream>>>(hbuf, conv_W + (size_t)i * HH * HH,
                                                  stA, stC, nullptr, zbuf, NN, 0);
        k_prop_init<<<gNW, 256, 0, stream>>>(pbuf, zbuf, dinvO, NN);
        k_prop_edges<<<gEW, 256, 0, stream>>>(pbuf, zbuf, dinvO, row, col, nullptr, EE);
        k_bias_relu<<<gNH, 256, 0, stream>>>(hbuf, pbuf, conv_b + i * HH, NN);
    }

    // 3) attention: node gating (xc=zbuf, xo=pbuf) and edge weights
    k_natt_split<<<(NN + 7) / 8, 256, 0, stream>>>(hbuf, W_natt, b_natt, zbuf, pbuf, NN);
    k_eatt<<<(EE + 7) / 8, 256, 0, stream>>>(hbuf, row, col, W_eatt, b_eatt, ea0, ea1, EE);

    // 4) causal branch: xc conv with edge weights ea0
    k_colstats<<<HH, 256, 0, stream>>>(zbuf, NN, bnc_g, bnc_b, stA, stC);
    k_gemm_bn_wmma<<<gRows, 256, 0, stream>>>(zbuf, Wc, stA, stC, nullptr, hbuf, NN, 0);
    k_deg_init<<<gN, 256, 0, stream>>>(deg, NN);
    k_deg_edges<<<gE, 256, 0, stream>>>(deg, row, ea0, EE);
    k_dinv<<<gN, 256, 0, stream>>>(deg, dinvA, NN);
    k_prop_init<<<gNW, 256, 0, stream>>>(zbuf, hbuf, dinvA, NN);
    k_prop_edges<<<gEW, 256, 0, stream>>>(zbuf, hbuf, dinvA, row, col, ea0, EE);
    k_bias_relu<<<gNH, 256, 0, stream>>>(zbuf, zbuf, bc, NN);
    k_zero<<<(GG * HH + 255) / 256, 256, 0, stream>>>(xcp, GG * HH);
    k_pool<<<gNW, 256, 0, stream>>>(xcp, zbuf, batch, NN);

    // 5) object branch: xo conv with edge weights ea1
    k_colstats<<<HH, 256, 0, stream>>>(pbuf, NN, bno_g, bno_b, stA, stC);
    k_gemm_bn_wmma<<<gRows, 256, 0, stream>>>(pbuf, Wo, stA, stC, nullptr, hbuf, NN, 0);
    k_deg_init<<<gN, 256, 0, stream>>>(deg, NN);
    k_deg_edges<<<gE, 256, 0, stream>>>(deg, row, ea1, EE);
    k_dinv<<<gN, 256, 0, stream>>>(deg, dinvA, NN);
    k_prop_init<<<gNW, 256, 0, stream>>>(pbuf, hbuf, dinvA, NN);
    k_prop_edges<<<gEW, 256, 0, stream>>>(pbuf, hbuf, dinvA, row, col, ea1, EE);
    k_bias_relu<<<gNH, 256, 0, stream>>>(pbuf, pbuf, bo, NN);
    k_zero<<<(GG * HH + 255) / 256, 256, 0, stream>>>(xop, GG * HH);
    k_pool<<<gNW, 256, 0, stream>>>(xop, pbuf, batch, NN);

    // 6) xc + xo
    k_add<<<(GG * HH + 255) / 256, 256, 0, stream>>>(xsum, xcp, xop, GG * HH);

    // 7) three readout heads -> log_softmax into d_out
    const float* ins[3] = { xcp, xop, xsum };
    for (int i = 0; i < 3; ++i) {
        k_colstats<<<HH, 256, 0, stream>>>(ins[i], GG, ro_bn1_g + i * HH, ro_bn1_b + i * HH, stA, stC);
        k_gemm_bn_wmma<<<1, 256, 0, stream>>>(ins[i], ro_W1 + (size_t)i * HH * HH,
                                              stA, stC, ro_b1 + i * HH, tbuf, GG, 1);
        k_colstats<<<HH, 256, 0, stream>>>(tbuf, GG, ro_bn2_g + i * HH, ro_bn2_b + i * HH, stA2, stC2);
        k_gemm_small<<<(GG * CC + 255) / 256, 256, 0, stream>>>(tbuf, ro_W2 + (size_t)i * HH * CC,
                                                                stA2, stC2, ro_b2 + i * CC, logit);
        k_logsoftmax<<<1, 128, 0, stream>>>(logit, out + (size_t)i * GG * CC);
    }
}
